// TinyVideoSwinEncoder_14783277433307
// MI455X (gfx1250) — matrix operand order repaired and checked by reading, not compiled, not run on hardware
//
#include <hip/hip_runtime.h>
#include <hip/hip_bf16.h>
#include <math.h>

// ---------------- problem constants ----------------
#define BATCH   32
#define DDIM    10
#define HDIM    30
#define WDIM    40
#define CH_IN   11
#define DM      128
#define TOK     (BATCH*DDIM*HDIM*WDIM)      // 384000
#define NWIN_B  96                          // 2*6*8 windows per batch sample
#define WINS    (BATCH*NWIN_B)              // 3072
#define NTOK    125                         // 5*5*5 tokens per window
#define VIS     (DDIM*CH_IN*HDIM*WDIM)      // 132000
#define AUX     20
#define OBS_ROW (VIS+AUX)                   // 132020
#define OUTN    256
#define ATT_SCALE 0.08838834764831845f      // 128^-0.5

// ---------------- WMMA plumbing ----------------
typedef __attribute__((ext_vector_type(16))) __bf16 v16bf;
typedef __attribute__((ext_vector_type(8)))  float  v8f;

union FragU { uint4 q[2]; v16bf v; };

__device__ __forceinline__ unsigned short f2bf(float f) {
  unsigned int u = __builtin_bit_cast(unsigned int, f);
  u += 0x7FFFu + ((u >> 16) & 1u);            // round-to-nearest-even
  return (unsigned short)(u >> 16);
}

// A fragment, 16x32 bf16, row-major tile with row stride `ld` (ushorts).
// ISA layout: lane holds row (lane&15); K chunks [kh,kh+8) and [16+kh,16+kh+8), kh=(lane>>4)*8.
__device__ __forceinline__ v16bf frag_a16(const unsigned short* tile, int ld, int lane) {
  int r = lane & 15, kh = (lane >> 4) << 3;
  FragU f;
  f.q[0] = *(const uint4*)(tile + r * ld + kh);
  f.q[1] = *(const uint4*)(tile + r * ld + 16 + kh);
  return f.v;
}
// B fragment, 32x16 bf16, sourced from an [n][k] tile with row stride `ld`.
// ISA layout: lane holds column N=(lane&15); K chunks [kh,kh+8) and [16+kh,16+kh+8), kh=(lane>>4)*8.
__device__ __forceinline__ v16bf frag_b16(const unsigned short* tile, int ld, int lane) {
  int n = lane & 15, ks = (lane >> 4) << 3;
  const unsigned short* p = tile + n * ld + ks;
  FragU f;
  f.q[0] = *(const uint4*)(p);
  f.q[1] = *(const uint4*)(p + 16);
  return f.v;
}
__device__ __forceinline__ v8f wmma_bf16(v16bf a, v16bf b, v8f c) {
  return __builtin_amdgcn_wmma_f32_16x16x32_bf16(false, a, false, b, (short)0, c, false, false);
}
__device__ __forceinline__ v8f v8f_zero() {
  v8f z;
  #pragma unroll
  for (int i = 0; i < 8; ++i) z[i] = 0.0f;
  return z;
}

// ---------------- CDNA5 async global->LDS copy (ASYNCcnt path) ----------------
// GV mode: LDS address taken from the VDST VGPR; global addr from 64-bit VGPR pair.
__device__ __forceinline__ void async_copy16(const unsigned short* g, unsigned short* l) {
  unsigned long long ga = (unsigned long long)(uintptr_t)g;
  unsigned int la = (unsigned int)(uintptr_t)l;       // low 32 bits = LDS offset
  asm volatile("global_load_async_to_lds_b128 %0, %1, off"
               :: "v"(la), "v"(ga) : "memory");
}
#define WAIT_ASYNC(n) asm volatile("s_wait_asynccnt " #n ::: "memory")

// ---------------- fp32 -> bf16 convert (elementwise) ----------------
__global__ __launch_bounds__(256) void k_cvt(const float* __restrict__ s,
                                             unsigned short* __restrict__ d, int n) {
  int i = blockIdx.x * 256 + threadIdx.x;
  if (i < n) d[i] = f2bf(s[i]);
}
// fp32 [K][N] -> bf16 [N][K] (transpose once so GEMM B-tiles are linear copies)
__global__ __launch_bounds__(256) void k_cvt_t(const float* __restrict__ s,
                                               unsigned short* __restrict__ d,
                                               int K, int N) {
  int i = blockIdx.x * 256 + threadIdx.x;
  if (i < K * N) {
    int k = i / N, n = i % N;
    d[(size_t)n * K + k] = f2bf(s[i]);
  }
}

// ---------------- patch embed: obs -> x fp32 [TOK][128] ----------------
__global__ __launch_bounds__(256) void k_patch(const float* __restrict__ obs,
                                               const float* __restrict__ pw,
                                               const float* __restrict__ pb,
                                               float* __restrict__ x) {
  int t = blockIdx.x * 8 + (threadIdx.x >> 5);          // token, wave-per-token
  int lane = threadIdx.x & 31;
  int b = t / 12000, rem = t % 12000;
  int d = rem / 1200, h = (rem / 40) % 30, w = rem % 40;
  int c = lane * 4;
  float acc0 = pb[c], acc1 = pb[c+1], acc2 = pb[c+2], acc3 = pb[c+3];
  const float* ob = obs + (size_t)b * OBS_ROW;
  #pragma unroll
  for (int ch = 0; ch < CH_IN; ++ch) {
    float v = ob[((d * CH_IN + ch) * HDIM + h) * WDIM + w];
    const float* pwr = pw + ch * DM + c;
    acc0 += v * pwr[0]; acc1 += v * pwr[1]; acc2 += v * pwr[2]; acc3 += v * pwr[3];
  }
  float4 o = {acc0, acc1, acc2, acc3};
  *(float4*)(x + (size_t)t * DM + c) = o;
}

// ---------------- LayerNorm helper (wave per token, 4 ch/lane) ----------------
__device__ __forceinline__ void ln_load(const float* src, int lane, float4& v,
                                        float& mean, float& inv) {
  v = *(const float4*)(src + lane * 4);
  float s = v.x + v.y + v.z + v.w;
  #pragma unroll
  for (int o = 16; o > 0; o >>= 1) s += __shfl_xor(s, o, 32);
  mean = s * (1.0f / 128.0f);
  float dx = v.x - mean, dy = v.y - mean, dz = v.z - mean, dw = v.w - mean;
  float q = dx*dx + dy*dy + dz*dz + dw*dw;
  #pragma unroll
  for (int o = 16; o > 0; o >>= 1) q += __shfl_xor(q, o, 32);
  inv = rsqrtf(q * (1.0f / 128.0f) + 1e-5f);
}

// LN + roll + window-partition -> bf16 xw [WINS*NTOK][128]
__global__ __launch_bounds__(256) void k_ln_window(const float* __restrict__ x,
                                                   const float* __restrict__ lw,
                                                   const float* __restrict__ lb,
                                                   unsigned short* __restrict__ xw,
                                                   int sd, int sh, int sw) {
  int gt = blockIdx.x * 8 + (threadIdx.x >> 5);   // window-token id
  int lane = threadIdx.x & 31;
  int win = gt / NTOK, i = gt % NTOK;
  int b  = win / NWIN_B, wr = win % NWIN_B;
  int wd = wr / 48, wh = (wr / 8) % 6, ww = wr % 8;
  int td = i / 25, th = (i / 5) % 5, tw = i % 5;
  int d = (wd * 5 + td + sd) % DDIM;
  int h = (wh * 5 + th + sh) % HDIM;
  int w = (ww * 5 + tw + sw) % WDIM;
  const float* src = x + (size_t)(((b * DDIM + d) * HDIM + h) * WDIM + w) * DM;
  float4 v; float mean, inv;
  ln_load(src, lane, v, mean, inv);
  int c = lane * 4;
  ushort4 o;
  o.x = f2bf((v.x - mean) * inv * lw[c]   + lb[c]);
  o.y = f2bf((v.y - mean) * inv * lw[c+1] + lb[c+1]);
  o.z = f2bf((v.z - mean) * inv * lw[c+2] + lb[c+2]);
  o.w = f2bf((v.w - mean) * inv * lw[c+3] + lb[c+3]);
  *(ushort4*)(xw + (size_t)gt * DM + c) = o;
}

// plain LN (identity token order) -> bf16 [TOK][128]
__global__ __launch_bounds__(256) void k_ln_tok(const float* __restrict__ x,
                                                const float* __restrict__ lw,
                                                const float* __restrict__ lb,
                                                unsigned short* __restrict__ out) {
  int t = blockIdx.x * 8 + (threadIdx.x >> 5);
  int lane = threadIdx.x & 31;
  const float* src = x + (size_t)t * DM;
  float4 v; float mean, inv;
  ln_load(src, lane, v, mean, inv);
  int c = lane * 4;
  ushort4 o;
  o.x = f2bf((v.x - mean) * inv * lw[c]   + lb[c]);
  o.y = f2bf((v.y - mean) * inv * lw[c+1] + lb[c+1]);
  o.z = f2bf((v.z - mean) * inv * lw[c+2] + lb[c+2]);
  o.w = f2bf((v.w - mean) * inv * lw[c+3] + lb[c+3]);
  *(ushort4*)(out + (size_t)t * DM + c) = o;
}

// ---------------- WMMA GEMM: C = A(bf16 MxK) @ B + bias; B pre-transposed [N][K] ----------------
// Double-buffered LDS, tiles filled by GLOBAL_LOAD_ASYNC_TO_LDS_B128 (ASYNCcnt).
// MODE 0: store bf16; MODE 1: exact GELU then bf16; MODE 2: fp32 += (residual)
template <int MODE>
__global__ __launch_bounds__(256) void k_gemm(const unsigned short* __restrict__ A,
                                              const unsigned short* __restrict__ Bt,
                                              const float* __restrict__ bias,
                                              void* __restrict__ outp,
                                              int M, int N, int K) {
  __shared__ unsigned short As[2][128 * 32];   // [m][k]
  __shared__ unsigned short Bs[2][128 * 32];   // [n][k]
  int tid = threadIdx.x, lane = tid & 31, wv = tid >> 5;
  int bm = blockIdx.y * 128, bn = blockIdx.x * 128;
  int r = tid >> 1, cc = (tid & 1) * 16;        // each thread owns a 16-ushort half of a 32-ushort row
  const unsigned short* ga = A  + (size_t)(bm + r) * K + cc;
  const unsigned short* gb = Bt + (size_t)(bn + r) * K + cc;

  v8f acc[8];
  #pragma unroll
  for (int t = 0; t < 8; ++t) acc[t] = v8f_zero();

  const int NK = K / 32;
  // prologue: k-step 0 into buffer 0 (4 async b128 per lane)
  async_copy16(ga,     &As[0][r * 32 + cc]);
  async_copy16(ga + 8, &As[0][r * 32 + cc + 8]);
  async_copy16(gb,     &Bs[0][r * 32 + cc]);
  async_copy16(gb + 8, &Bs[0][r * 32 + cc + 8]);

  for (int ks = 0; ks < NK; ++ks) {
    int cur = ks & 1;
    if (ks + 1 < NK) {
      int nxt = cur ^ 1, k0 = (ks + 1) * 32;
      async_copy16(ga + k0,     &As[nxt][r * 32 + cc]);
      async_copy16(ga + k0 + 8, &As[nxt][r * 32 + cc + 8]);
      async_copy16(gb + k0,     &Bs[nxt][r * 32 + cc]);
      async_copy16(gb + k0 + 8, &Bs[nxt][r * 32 + cc + 8]);
      WAIT_ASYNC(0x4);                 // older 4 (current buffer) done; next 4 in flight
    } else {
      WAIT_ASYNC(0x0);
    }
    __syncthreads();
    v16bf a = frag_a16(&As[cur][(wv * 16) * 32], 32, lane);
    #pragma unroll
    for (int nt = 0; nt < 8; ++nt) {
      v16bf b = frag_b16(&Bs[cur][(nt * 16) * 32], 32, lane);
      acc[nt] = wmma_bf16(a, b, acc[nt]);
    }
    __syncthreads();                   // all waves done before buffer is refilled
  }

  int m0 = bm + wv * 16 + ((lane >> 4) << 3);
  int nlane = lane & 15;
  #pragma unroll
  for (int nt = 0; nt < 8; ++nt) {
    int n = bn + nt * 16 + nlane;
    float bval = bias[n];
    #pragma unroll
    for (int rr = 0; rr < 8; ++rr) {
      int m = m0 + rr;
      float val = acc[nt][rr] + bval;
      if (MODE == 1) val = 0.5f * val * (1.0f + erff(val * 0.70710678118654752f));
      if (MODE == 2) ((float*)outp)[(size_t)m * N + n] += val;
      else ((unsigned short*)outp)[(size_t)m * N + n] = f2bf(val);
    }
  }
}

// ---------------- per-window attention (one workgroup = one window) ----------------
__device__ __forceinline__ int region3(int v, int L) {        // [0,L-5) / [L-5,L-2) / [L-2,L)
  return (v < L - 5) ? 0 : ((v < L - 2) ? 1 : 2);
}

__global__ __launch_bounds__(256) void k_attn(const unsigned short* __restrict__ qkv,
                                              const float* __restrict__ rpb,
                                              unsigned short* __restrict__ out,
                                              int shifted) {
  extern __shared__ unsigned short smem[];
  unsigned short* Qs = smem;                       // [tok][ch] (later reused for P)
  unsigned short* Ks = smem + 128 * 128;           // [tok][ch] (== B-layout for Q@K^T)
  unsigned short* Vt = smem + 2 * 128 * 128;       // [ch][tok]
  float* Sf = (float*)(smem + 3 * 128 * 128);      // scores fp32 128x128

  int tid = threadIdx.x, lane = tid & 31, wv = tid >> 5;
  int win = blockIdx.x;
  size_t base = (size_t)win * NTOK * 384;

  // ---- stage Q,K via async global->LDS; V transposed manually; pad >=125 with zeros ----
  {
    int r = tid >> 1, c0 = (tid & 1) * 64;
    if (r < NTOK) {
      const unsigned short* q = qkv + base + (size_t)r * 384 + c0;
      const unsigned short* k = q + 128;
      #pragma unroll
      for (int j = 0; j < 64; j += 8) {
        async_copy16(q + j, Qs + r * 128 + c0 + j);
        async_copy16(k + j, Ks + r * 128 + c0 + j);
      }
    } else {
      uint4 z = {0u, 0u, 0u, 0u};
      #pragma unroll
      for (int j = 0; j < 64; j += 8) {
        *(uint4*)(Qs + r * 128 + c0 + j) = z;
        *(uint4*)(Ks + r * 128 + c0 + j) = z;
      }
    }
    #pragma unroll
    for (int ccx = 0; ccx < 64; ++ccx) {
      unsigned short v = (r < NTOK) ? qkv[base + (size_t)r * 384 + 256 + c0 + ccx]
                                    : (unsigned short)0;
      Vt[(c0 + ccx) * 128 + r] = v;                 // transpose: [ch][tok]
    }
  }
  WAIT_ASYNC(0x0);
  __syncthreads();

  // ---- S = Q @ K^T ----
  v8f acc[8];
  #pragma unroll
  for (int t = 0; t < 8; ++t) acc[t] = v8f_zero();
  #pragma unroll
  for (int k0 = 0; k0 < 128; k0 += 32) {
    v16bf a = frag_a16(Qs + (wv * 16) * 128 + k0, 128, lane);
    #pragma unroll
    for (int nt = 0; nt < 8; ++nt) {
      v16bf b = frag_b16(Ks + (nt * 16) * 128 + k0, 128, lane);
      acc[nt] = wmma_bf16(a, b, acc[nt]);
    }
  }

  // ---- score epilogue: scale + rel-pos bias + causal + boundary mask ----
  int wr = win % NWIN_B;
  int wd = wr / 48, wh = (wr / 8) % 6, ww = wr % 8;
  int i0 = wv * 16 + ((lane >> 4) << 3);
  int j0 = lane & 15;
  #pragma unroll
  for (int nt = 0; nt < 8; ++nt) {
    #pragma unroll
    for (int rr = 0; rr < 8; ++rr) {
      int i = i0 + rr, j = j0 + nt * 16;
      float s;
      if (i < NTOK && j < NTOK) {
        s = acc[nt][rr] * ATT_SCALE;
        int tdi = i / 25, thi = (i / 5) % 5, twi = i % 5;
        int tdj = j / 25, thj = (j / 5) % 5, twj = j % 5;
        s += rpb[(tdi - tdj + 4) * 81 + (thi - thj + 4) * 9 + (twi - twj + 4)];
        if (tdi < tdj) s -= 100.0f;                               // temporal causal
        if (shifted) {
          int li = region3(wd * 5 + tdi, DDIM) * 9 + region3(wh * 5 + thi, HDIM) * 3
                 + region3(ww * 5 + twi, WDIM);
          int lj = region3(wd * 5 + tdj, DDIM) * 9 + region3(wh * 5 + thj, HDIM) * 3
                 + region3(ww * 5 + twj, WDIM);
          if (li != lj) s -= 100.0f;                              // boundary mask
        }
      } else {
        s = -1e30f;
      }
      Sf[i * 128 + j] = s;
    }
  }
  __syncthreads();

  // ---- softmax rows -> P (bf16) into Qs ----
  if (tid < 128) {
    float mx = -1e30f;
    for (int j = 0; j < NTOK; ++j) mx = fmaxf(mx, Sf[tid * 128 + j]);
    float sum = 0.0f;
    for (int j = 0; j < NTOK; ++j) {
      float e = __expf(Sf[tid * 128 + j] - mx);
      Sf[tid * 128 + j] = e;
      sum += e;
    }
    float rinv = 1.0f / sum;
    for (int j = 0; j < NTOK; ++j) Qs[tid * 128 + j] = f2bf(Sf[tid * 128 + j] * rinv);
    Qs[tid * 128 + 125] = 0; Qs[tid * 128 + 126] = 0; Qs[tid * 128 + 127] = 0;
  }
  __syncthreads();

  // ---- O = P @ V ----
  v8f acc2[8];
  #pragma unroll
  for (int t = 0; t < 8; ++t) acc2[t] = v8f_zero();
  #pragma unroll
  for (int k0 = 0; k0 < 128; k0 += 32) {
    v16bf a = frag_a16(Qs + (wv * 16) * 128 + k0, 128, lane);
    #pragma unroll
    for (int nt = 0; nt < 8; ++nt) {
      v16bf b = frag_b16(Vt + (nt * 16) * 128 + k0, 128, lane);
      acc2[nt] = wmma_bf16(a, b, acc2[nt]);
    }
  }
  #pragma unroll
  for (int nt = 0; nt < 8; ++nt) {
    #pragma unroll
    for (int rr = 0; rr < 8; ++rr) {
      int i = i0 + rr, ch = j0 + nt * 16;
      if (i < NTOK)
        out[((size_t)win * NTOK + i) * DM + ch] = f2bf(acc2[nt][rr]);
    }
  }
}

// ---------------- window-reverse + roll-back + residual add into x ----------------
__global__ __launch_bounds__(256) void k_scatter(const unsigned short* __restrict__ proj,
                                                 float* __restrict__ x,
                                                 int sd, int sh, int sw) {
  size_t idx = (size_t)blockIdx.x * 256 + threadIdx.x;   // over TOK*128
  int c = (int)(idx & 127);
  int gt = (int)(idx >> 7);
  int win = gt / NTOK, i = gt % NTOK;
  int b  = win / NWIN_B, wr = win % NWIN_B;
  int wd = wr / 48, wh = (wr / 8) % 6, ww = wr % 8;
  int td = i / 25, th = (i / 5) % 5, tw = i % 5;
  int d = (wd * 5 + td + sd) % DDIM;
  int h = (wh * 5 + th + sh) % HDIM;
  int w = (ww * 5 + tw + sw) % WDIM;
  unsigned int bv = (unsigned int)proj[(size_t)gt * DM + c] << 16;
  x[(size_t)(((b * DDIM + d) * HDIM + h) * WDIM + w) * DM + c] +=
      __builtin_bit_cast(float, bv);
}

// ---------------- final LN + mean pool (atomic) + head ----------------
__global__ __launch_bounds__(256) void k_zero(float* __restrict__ p, int n) {
  int i = blockIdx.x * 256 + threadIdx.x;
  if (i < n) p[i] = 0.0f;
}

__global__ __launch_bounds__(256) void k_pool(const float* __restrict__ x,
                                              const float* __restrict__ lw,
                                              const float* __restrict__ lb,
                                              float* __restrict__ pool) {
  int t = blockIdx.x * 8 + (threadIdx.x >> 5);
  int lane = threadIdx.x & 31;
  int b = t / 12000;
  const float* src = x + (size_t)t * DM;
  float4 v; float mean, inv;
  ln_load(src, lane, v, mean, inv);
  int c = lane * 4;
  const float sc = 1.0f / 12000.0f;
  atomicAdd(pool + b * DM + c,     ((v.x - mean) * inv * lw[c]   + lb[c])   * sc);
  atomicAdd(pool + b * DM + c + 1, ((v.y - mean) * inv * lw[c+1] + lb[c+1]) * sc);
  atomicAdd(pool + b * DM + c + 2, ((v.z - mean) * inv * lw[c+2] + lb[c+2]) * sc);
  atomicAdd(pool + b * DM + c + 3, ((v.w - mean) * inv * lw[c+3] + lb[c+3]) * sc);
}

__global__ __launch_bounds__(256) void k_head(const float* __restrict__ pool,
                                              const float* __restrict__ obs,
                                              const float* __restrict__ ow,
                                              const float* __restrict__ ob,
                                              float* __restrict__ out) {
  int b = blockIdx.x, o = threadIdx.x;      // 32 blocks x 256 threads
  float s = ob[o];
  const float* pr = pool + b * DM;
  for (int f = 0; f < DM; ++f) s += pr[f] * ow[f * OUTN + o];
  const float* aux = obs + (size_t)b * OBS_ROW + VIS;
  for (int j = 0; j < AUX; ++j) s += aux[j] * ow[(DM + j) * OUTN + o];
  out[b * OUTN + o] = s;
}

// ---------------- launch ----------------
extern "C" void kernel_launch(void* const* d_in, const int* in_sizes, int n_in,
                              void* d_out, int out_size, void* d_ws, size_t ws_size,
                              hipStream_t stream) {
  (void)in_sizes; (void)n_in; (void)out_size; (void)ws_size;
  const float* obs    = (const float*)d_in[0];
  const float* patchw = (const float*)d_in[1];
  const float* patchb = (const float*)d_in[2];
  const float* n1w    = (const float*)d_in[3];
  const float* n1b    = (const float*)d_in[4];
  const float* qkvw   = (const float*)d_in[5];
  const float* qkvb   = (const float*)d_in[6];
  const float* rpb    = (const float*)d_in[7];
  const float* projw  = (const float*)d_in[8];
  const float* projb  = (const float*)d_in[9];
  const float* n2w    = (const float*)d_in[10];
  const float* n2b    = (const float*)d_in[11];
  const float* w1     = (const float*)d_in[12];
  const float* b1     = (const float*)d_in[13];
  const float* w2     = (const float*)d_in[14];
  const float* b2     = (const float*)d_in[15];
  const float* nfw    = (const float*)d_in[16];
  const float* nfb    = (const float*)d_in[17];
  const float* outw   = (const float*)d_in[18];
  const float* outb   = (const float*)d_in[19];

  // workspace layout (byte offsets)
  char* ws = (char*)d_ws;
  const size_t X_OFF    = 0;                                    // fp32 x  [TOK][128]
  const size_t BUFA_OFF = X_OFF    + (size_t)TOK * DM * 4;      // bf16    [TOK][128]
  const size_t BUFB_OFF = BUFA_OFF + (size_t)TOK * DM * 2;      // bf16    [TOK][512] (qkv/mlp)
  const size_t ATTN_OFF = BUFB_OFF + (size_t)TOK * 512 * 2;     // bf16    [TOK][128]
  const size_t POOL_OFF = ATTN_OFF + (size_t)TOK * DM * 2;      // fp32    [32][128]
  const size_t WQKV_OFF = POOL_OFF + (size_t)BATCH * DM * 4;    // bf16 weights, transposed [N][K]
  const size_t WPRJ_OFF = WQKV_OFF + (size_t)2 * DM * 384 * 2;
  const size_t WM1_OFF  = WPRJ_OFF + (size_t)2 * DM * DM * 2;
  const size_t WM2_OFF  = WM1_OFF  + (size_t)2 * DM * 512 * 2;

  float*          x     = (float*)(ws + X_OFF);
  unsigned short* bufA  = (unsigned short*)(ws + BUFA_OFF);
  unsigned short* bufB  = (unsigned short*)(ws + BUFB_OFF);
  unsigned short* attnb = (unsigned short*)(ws + ATTN_OFF);
  float*          pool  = (float*)(ws + POOL_OFF);
  unsigned short* wqkv  = (unsigned short*)(ws + WQKV_OFF);
  unsigned short* wprj  = (unsigned short*)(ws + WPRJ_OFF);
  unsigned short* wm1   = (unsigned short*)(ws + WM1_OFF);
  unsigned short* wm2   = (unsigned short*)(ws + WM2_OFF);

  // transpose-convert weights to bf16 [N][K] (per layer)
  for (int i = 0; i < 2; ++i) {
    k_cvt_t<<<(DM*384 + 255)/256, 256, 0, stream>>>(qkvw + (size_t)i*DM*384,
                                                    wqkv + (size_t)i*DM*384, DM, 384);
    k_cvt_t<<<(DM*DM  + 255)/256, 256, 0, stream>>>(projw + (size_t)i*DM*DM,
                                                    wprj + (size_t)i*DM*DM, DM, DM);
    k_cvt_t<<<(DM*512 + 255)/256, 256, 0, stream>>>(w1 + (size_t)i*DM*512,
                                                    wm1 + (size_t)i*DM*512, DM, 512);
    k_cvt_t<<<(512*DM + 255)/256, 256, 0, stream>>>(w2 + (size_t)i*512*DM,
                                                    wm2 + (size_t)i*512*DM, 512, DM);
  }

  const int TB = TOK / 8;  // 48000 wave-per-token blocks
  k_patch<<<TB, 256, 0, stream>>>(obs, patchw, patchb, x);

  const size_t ATTN_LDS = (size_t)3 * 128 * 128 * 2 + (size_t)128 * 128 * 4;  // 160 KB

  for (int i = 0; i < 2; ++i) {
    int sd = (i % 2) ? 2 : 0, sh = sd, sw = sd, shifted = (i % 2);
    k_ln_window<<<TB, 256, 0, stream>>>(x, n1w + i*DM, n1b + i*DM, bufA, sd, sh, sw);
    k_gemm<0><<<dim3(3, TOK/128), 256, 0, stream>>>(bufA, wqkv + (size_t)i*DM*384,
                                                    qkvb + i*384, bufB, TOK, 384, DM);
    k_attn<<<WINS, 256, ATTN_LDS, stream>>>(bufB, rpb + i*729, attnb, shifted);
    k_gemm<0><<<dim3(1, TOK/128), 256, 0, stream>>>(attnb, wprj + (size_t)i*DM*DM,
                                                    projb + i*DM, bufA, TOK, DM, DM);
    k_scatter<<<(TOK*DM)/256, 256, 0, stream>>>(bufA, x, sd, sh, sw);
    k_ln_tok<<<TB, 256, 0, stream>>>(x, n2w + i*DM, n2b + i*DM, bufA);
    k_gemm<1><<<dim3(4, TOK/128), 256, 0, stream>>>(bufA, wm1 + (size_t)i*DM*512,
                                                    b1 + i*512, bufB, TOK, 512, DM);
    k_gemm<2><<<dim3(1, TOK/128), 256, 0, stream>>>(bufB, wm2 + (size_t)i*512*DM,
                                                    b2 + i*DM, x, TOK, DM, 512);
  }

  k_zero<<<(BATCH*DM + 255)/256, 256, 0, stream>>>(pool, BATCH*DM);
  k_pool<<<TB, 256, 0, stream>>>(x, nfw, nfb, pool);
  k_head<<<BATCH, 256, 0, stream>>>(pool, obs, outw, outb, (float*)d_out);
}
